// ZeroHead_47407849013811
// MI455X (gfx1250) — matrix operand order repaired and checked by reading, not compiled, not run on hardware
//
#include <hip/hip_runtime.h>
#include <math.h>

// ZeroHead (GFL/NanoDet head) as fused implicit-GEMM conv + epilogue.
// One kernel per FPN level:  D[M=N*H*W, 149] = im2col(x)[M, 9C] * W^T[9C, 149]
// f16 WMMA (v_wmma_f32_16x16x32_f16), f32 accumulate, LDS ping-pong pipeline.
// Epilogue: sigmoid(cls), per-side softmax-over-17-bins + integral, bbox decode.

typedef __attribute__((ext_vector_type(16))) _Float16 v16h;
typedef __attribute__((ext_vector_type(8)))  float    v8f;
typedef __attribute__((ext_vector_type(4)))  unsigned int v4u;

#define NUM_CLASSES 80
#define CLS_OUT     81          // 80 classes + 1 (dropped at the end)
#define REG_OUT     68          // 4 * 17 bins
#define NTOT        149         // 81 + 68 fused output channels
#define NTILES      10          // 160 padded channels / 16
#define A_TOTAL     8400        // 6400 + 1600 + 400 anchors

union FragU { v16h h; v4u u[2]; };
union PackA { _Float16 h[16]; v4u u[2]; };
union PackB { _Float16 h[4];  unsigned long long q; };

template<int C, int H, int W>
__global__ __launch_bounds__(256)
void zerohead_level_kernel(const float* __restrict__ x,
                           const float* __restrict__ wc, const float* __restrict__ bc,
                           const float* __restrict__ wr, const float* __restrict__ br,
                           const float* __restrict__ scales, int level,
                           int a_off, int stride,
                           float* __restrict__ out_cls, float* __restrict__ out_box)
{
    constexpr int K  = C * 9;
    constexpr int HW = H * W;
    constexpr int KT = K / 32;   // divisible at every level

    // GEMM tiles (double-buffered) union'd with epilogue staging buffers.
    struct GemmBufs { _Float16 A[2][128 * 32]; _Float16 B[2][160 * 32]; };
    struct EpiBufs  { float reg[128][REG_OUT + 4]; float dist[128][4]; };
    __shared__ union SmemU { GemmBufs g; EpiBufs e; } smem;

    const int tid   = threadIdx.x;
    const int wave  = tid >> 5;
    const int lane  = tid & 31;
    const int lhalf = lane >> 4;
    const int l16   = lane & 15;
    const long mBase = (long)blockIdx.x * 128;

    // ---- A staging geometry: thread owns row aRow, one half-row of 16 k's ----
    const int  aRow     = tid >> 1;              // 0..127
    const int  aColBase = (tid & 1) * 16;        // 0 or 16
    const long mG   = mBase + aRow;
    const int  b_a  = (int)(mG / HW);
    const int  hw_a = (int)(mG % HW);
    const int  hy_a = hw_a / W;
    const int  wx_a = hw_a % W;
    const float* xbase = x + (long)b_a * C * HW;

    float  aReg[16];
    float4 bReg[5];

    // fetch im2col values for K-step kt into registers (division hoisted,
    // (c,dy,dx) walked incrementally across the unrolled 16 elements)
    auto fetchA = [&](int kt) {
        const int kb0 = kt * 32 + aColBase;
        int c  = kb0 / 9;                        // one div per thread per K-step
        int rs = kb0 - c * 9;
        int dy = rs / 3 - 1;
        int dx = rs - (rs / 3) * 3 - 1;
        long cOff = (long)c * HW;
        #pragma unroll
        for (int kk = 0; kk < 16; ++kk) {
            const int iy = hy_a + dy, ix = wx_a + dx;
            float v = 0.0f;
            if ((unsigned)iy < (unsigned)H && (unsigned)ix < (unsigned)W)
                v = xbase[cOff + iy * W + ix];
            aReg[kk] = v;
            ++dx;
            if (dx > 1) { dx = -1; ++dy; }
            if (dy > 1) { dy = -1; cOff += HW; }
        }
    };
    auto storeA = [&](int buf) {
        PackA p;
        #pragma unroll
        for (int kk = 0; kk < 16; ++kk) p.h[kk] = (_Float16)aReg[kk];
        v4u* dst = (v4u*)&smem.g.A[buf][aRow * 32 + aColBase];
        dst[0] = p.u[0];                          // 2x ds_store_b128
        dst[1] = p.u[1];
    };
    // fused [cls | reg | zero-pad] weights; K-contiguous -> float4 global loads
    auto fetchB = [&](int kt) {
        const int kbase = kt * 32;
        #pragma unroll
        for (int j = 0; j < 5; ++j) {
            const int e  = j * 256 + tid;        // 1280 groups of 4 over 160x32
            const int n  = e >> 3;
            const int k4 = (e & 7) * 4;
            float4 v = make_float4(0.f, 0.f, 0.f, 0.f);
            if (n < CLS_OUT)   v = *(const float4*)&wc[(long)n * K + kbase + k4];
            else if (n < NTOT) v = *(const float4*)&wr[(long)(n - CLS_OUT) * K + kbase + k4];
            bReg[j] = v;
        }
    };
    auto storeB = [&](int buf) {
        #pragma unroll
        for (int j = 0; j < 5; ++j) {
            const int e  = j * 256 + tid;
            const int n  = e >> 3;
            const int k4 = (e & 7) * 4;
            PackB p;
            p.h[0] = (_Float16)bReg[j].x; p.h[1] = (_Float16)bReg[j].y;
            p.h[2] = (_Float16)bReg[j].z; p.h[3] = (_Float16)bReg[j].w;
            *(unsigned long long*)&smem.g.B[buf][n * 32 + k4] = p.q;  // ds_store_b64
        }
    };

    v8f acc[NTILES] = {};
    auto compute = [&](int buf) {
        // A frag (ISA 16-bit A 16x32): m = lane%16; two 16B chunks by lane-half
        FragU af;
        const _Float16* arow = &smem.g.A[buf][(wave * 16 + l16) * 32];
        af.u[0] = *(const v4u*)(arow + lhalf * 16);
        af.u[1] = *(const v4u*)(arow + lhalf * 16 + 8);
        #pragma unroll
        for (int nt = 0; nt < NTILES; ++nt) {
            // B frag (32x16): n = lane%16; K[lhalf*16 .. +15] contiguous (32B)
            FragU bfr;
            const _Float16* brow = &smem.g.B[buf][(nt * 16 + l16) * 32 + lhalf * 16];
            bfr.u[0] = *(const v4u*)(brow);
            bfr.u[1] = *(const v4u*)(brow + 8);
            acc[nt] = __builtin_amdgcn_wmma_f32_16x16x32_f16(
                false, af.h, false, bfr.h, (short)0, acc[nt], false, false);
        }
    };

    // ---- software-pipelined main loop: one barrier per K-step ----
    fetchA(0); fetchB(0); storeA(0); storeB(0);
    __syncthreads();
    for (int kt = 0; kt < KT; ++kt) {
        const int cur = kt & 1;
        const bool haveNext = (kt + 1 < KT);      // uniform branch (EXEC stays all-1s)
        if (haveNext) { fetchA(kt + 1); fetchB(kt + 1); }  // loads overlap WMMA below
        compute(cur);
        if (haveNext) { storeA(cur ^ 1); storeB(cur ^ 1); }
        __syncthreads();
    }

    // ---- epilogue A: cls sigmoid stores + stage reg logits to LDS ----
    const float scale = scales[level];
    float biasArr[NTILES];
    #pragma unroll
    for (int nt = 0; nt < NTILES; ++nt) {
        const int n = nt * 16 + l16;
        float b = 0.0f;
        if (n < CLS_OUT)   b = bc[n];
        else if (n < NTOT) b = br[n - CLS_OUT];
        biasArr[nt] = b;
    }
    #pragma unroll
    for (int r = 0; r < 8; ++r) {
        const int  mLocal = wave * 16 + lhalf * 8 + r;  // D layout: M = r (+8 hi half)
        const long mGlob  = mBase + mLocal;
        const int  bb = (int)(mGlob / HW);               // hoisted: 8x, not 80x
        const int  hw = (int)(mGlob % HW);
        float* clsRow = &out_cls[((long)bb * A_TOTAL + a_off + hw) * NUM_CLASSES];
        #pragma unroll
        for (int nt = 0; nt < NTILES; ++nt) {
            const int   n = nt * 16 + l16;               // D layout: N = lane%16
            const float v = acc[nt][r] + biasArr[nt];
            if (n < NUM_CLASSES) {
                clsRow[n] = 1.0f / (1.0f + __expf(-v));
            } else if (n >= CLS_OUT && n < NTOT) {
                smem.e.reg[mLocal][n - CLS_OUT] = v * scale;  // (conv+bias)*scale
            }
        }
    }
    __syncthreads();

    // ---- epilogue B: softmax over 17 bins + integral, per (pixel, side) ----
    #pragma unroll
    for (int it = 0; it < 2; ++it) {
        const int task = it * 256 + tid;         // 512 tasks = 128 px * 4 sides
        const int p = task >> 2;
        const int f = task & 3;
        const float* rp = &smem.e.reg[p][f * 17];
        float mx = rp[0];
        #pragma unroll
        for (int j = 1; j < 17; ++j) mx = fmaxf(mx, rp[j]);
        float sum = 0.0f, dot = 0.0f;
        #pragma unroll
        for (int j = 0; j < 17; ++j) {
            const float e = __expf(rp[j] - mx);
            sum += e;
            dot += e * (float)j;
        }
        smem.e.dist[p][f] = dot / sum;
    }
    __syncthreads();

    // ---- epilogue C: distance2bbox with grid priors ----
    if (tid < 128) {
        const long mGlob = mBase + tid;
        const int  bb = (int)(mGlob / HW);
        const int  hw = (int)(mGlob % HW);
        const float s  = (float)stride;
        const float px = (float)(hw % W) * s;
        const float py = (float)(hw / W) * s;
        float4 o;
        o.x = px - smem.e.dist[tid][0] * s;
        o.y = py - smem.e.dist[tid][1] * s;
        o.z = px + smem.e.dist[tid][2] * s;
        o.w = py + smem.e.dist[tid][3] * s;
        *(float4*)&out_box[((long)bb * A_TOTAL + a_off + hw) * 4] = o;
    }
}

extern "C" void kernel_launch(void* const* d_in, const int* in_sizes, int n_in,
                              void* d_out, int out_size, void* d_ws, size_t ws_size,
                              hipStream_t stream) {
    (void)in_sizes; (void)n_in; (void)out_size; (void)d_ws; (void)ws_size;
    // setup_inputs order: x0,x1,x2, then (wc,bc,wr,br) per level, then scales
    const float* x0 = (const float*)d_in[0];
    const float* x1 = (const float*)d_in[1];
    const float* x2 = (const float*)d_in[2];
    const float* wc0 = (const float*)d_in[3];  const float* bc0 = (const float*)d_in[4];
    const float* wr0 = (const float*)d_in[5];  const float* br0 = (const float*)d_in[6];
    const float* wc1 = (const float*)d_in[7];  const float* bc1 = (const float*)d_in[8];
    const float* wr1 = (const float*)d_in[9];  const float* br1 = (const float*)d_in[10];
    const float* wc2 = (const float*)d_in[11]; const float* bc2 = (const float*)d_in[12];
    const float* wr2 = (const float*)d_in[13]; const float* br2 = (const float*)d_in[14];
    const float* scales = (const float*)d_in[15];

    float* out     = (float*)d_out;
    float* out_cls = out;                                    // [32, 8400, 80]
    float* out_box = out + (long)32 * A_TOTAL * NUM_CLASSES; // [32, 8400, 4]

    // Level 0: C=128, 80x80, stride 8,  anchors [0,6400):    1600 M-tiles
    zerohead_level_kernel<128, 80, 80><<<1600, 256, 0, stream>>>(
        x0, wc0, bc0, wr0, br0, scales, 0, /*a_off=*/0, /*stride=*/8, out_cls, out_box);
    // Level 1: C=256, 40x40, stride 16, anchors [6400,8000): 400 M-tiles
    zerohead_level_kernel<256, 40, 40><<<400, 256, 0, stream>>>(
        x1, wc1, bc1, wr1, br1, scales, 1, /*a_off=*/6400, /*stride=*/16, out_cls, out_box);
    // Level 2: C=512, 20x20, stride 32, anchors [8000,8400): 100 M-tiles
    zerohead_level_kernel<512, 20, 20><<<100, 256, 0, stream>>>(
        x2, wc2, bc2, wr2, br2, scales, 2, /*a_off=*/8000, /*stride=*/32, out_cls, out_box);
}
